// GAT_33122787787016
// MI455X (gfx1250) — matrix operand order repaired and checked by reading, not compiled, not run on hardware
//
#include <hip/hip_runtime.h>
#include <hip/hip_bf16.h>

// ---------------------------------------------------------------------------
// GATv2 (3 layers) + classifier for MI455X (gfx1250, wave32, WMMA + TDM).
// - Dense GEMMs: v_wmma_f32_16x16x32_bf16 (bf16 in, f32 accum).
// - Layer GEMMs stage their weight slab in LDS via the Tensor Data Mover
//   (tensor_load_to_lds + s_wait_tensorcnt), fragments read via ds_load_b128.
// - Edge pipeline fused (never materializes E x 2048 edge-GEMM result);
//   xl/xr stored bf16 (40MB each) so edge gathers stay L2-resident (192MB L2).
// Workspace assumption: ws_size >= ~256MB.
// ---------------------------------------------------------------------------

typedef __bf16 bf16_t;
typedef bf16_t v16bf __attribute__((ext_vector_type(16)));
typedef float  v8f   __attribute__((ext_vector_type(8)));
typedef unsigned int v4u __attribute__((ext_vector_type(4)));
typedef int v8i __attribute__((ext_vector_type(8)));
typedef int v4i __attribute__((ext_vector_type(4)));

#define N_NODES 10000
#define N_EDGES 160000
#define D_NODE  64
#define D_EDGE  16
#define HID     32
#define HEADS   64
#define HC      2048   // HEADS*HID
#define NCLS    20
#define NEG_SLOPE 0.2f
#define EPS_SM  1e-16f

#define SLAB_COLS 512   // output columns staged per block in the TDM GEMM
#define NCT_L 4         // 16-col tiles per wave in the TDM GEMM (8 waves*64 = 512)

union Frag { v16bf v; uint4 u[2]; };

__device__ __forceinline__ unsigned f2ord(float f) {
  unsigned u = __float_as_uint(f);
  return (u >> 31) ? ~u : (u | 0x80000000u);
}
__device__ __forceinline__ float ord2f(unsigned o) {
  unsigned u = (o & 0x80000000u) ? (o & 0x7fffffffu) : ~o;
  return __uint_as_float(u);
}

// --------------------------- prep kernels ----------------------------------

__global__ __launch_bounds__(256)
void cvt_f32_bf16_kernel(const float* __restrict__ src, bf16_t* __restrict__ dst, int n) {
  int i = blockIdx.x * blockDim.x + threadIdx.x;
  if (i < n) dst[i] = (bf16_t)src[i];
}

// W: [K][NC] row-major f32  ->  WT: [NCpad][K] bf16 (zero-padded cols >= NC)
__global__ __launch_bounds__(256)
void transpose_w_kernel(const float* __restrict__ W, bf16_t* __restrict__ WT,
                        int K, int NC, int NCpad) {
  int i = blockIdx.x * blockDim.x + threadIdx.x;
  if (i >= K * NCpad) return;
  int col = i / K, k = i - col * K;
  float v = (col < NC) ? W[(size_t)k * NC + col] : 0.f;
  WT[(size_t)col * K + k] = (bf16_t)v;
}

// ----------------- WMMA GEMM with TDM-staged weights -----------------------
// C[M][NCols] = A[M][K](bf16) * W[K][NCols] + bias; WT is [NCols][K] bf16.
// One block = one 16-row tile x SLAB_COLS output columns. The block's weight
// slab (SLAB_COLS*K bf16, contiguous rows of WT) is DMA'd to LDS by the
// Tensor Data Mover; waves read B fragments with ds_load_b128.
// A 16x32 / B 32x16 fragments both use contiguous k = 16*hi + s (not ISA table order), f32 C/D.
__global__ __launch_bounds__(256)
void gemm_wmma_tdm_kernel(const bf16_t* __restrict__ A, const bf16_t* __restrict__ WT,
                          const float* __restrict__ bias, bf16_t* __restrict__ outB,
                          int M, int K, int NCols) {
  extern __shared__ bf16_t tileB[];        // SLAB_COLS * K bf16
  int wv   = threadIdx.x >> 5;
  int lane = threadIdx.x & 31;
  int colSlabs = NCols / SLAB_COLS;
  int mt   = blockIdx.x / colSlabs;
  int slab = blockIdx.x - mt * colSlabs;
  if (mt * 16 >= M) return;

  // ---- TDM: async contiguous copy of the weight slab into LDS (wave 0) ----
  if (wv == 0) {
    unsigned nElem = (unsigned)(SLAB_COLS * K);           // data_size units (2B)
    size_t   gaddr = (size_t)(WT + (size_t)slab * SLAB_COLS * K);
    unsigned laddr = (unsigned)(size_t)(&tileB[0]);       // LDS byte offset
    v4u g0;
    g0[0] = 1u;                                           // count=1 (valid D#)
    g0[1] = laddr;                                        // lds_addr
    g0[2] = (unsigned)(gaddr & 0xffffffffu);              // global_addr lo
    g0[3] = (unsigned)((gaddr >> 32) & 0x01ffffffu) | (2u << 30); // hi | type=2
    v8i g1;
    g1[0] = (int)(1u << 16);                              // data_size=1 (2B)
    g1[1] = (int)((nElem & 0xffffu) << 16);               // tensor_dim0 lo16
    g1[2] = (int)(((nElem >> 16) & 0xffffu) | (1u << 16));// tensor_dim0 hi | tensor_dim1=1
    g1[3] = (int)((nElem & 0xffffu) << 16);               // tile_dim0 = nElem
    g1[4] = 1;                                            // tile_dim1 = 1
    g1[5] = (int)nElem;                                   // tensor_dim0_stride
    g1[6] = 0;
    g1[7] = 0;
    v4i gz = {0, 0, 0, 0};
#if __clang_major__ >= 23
    v8i gz8 = {0, 0, 0, 0, 0, 0, 0, 0};
    __builtin_amdgcn_tensor_load_to_lds(g0, g1, gz, gz, gz8, 0);
#else
    __builtin_amdgcn_tensor_load_to_lds(g0, g1, gz, gz, 0);
#endif
    __builtin_amdgcn_s_wait_tensorcnt(0);
  }
  __syncthreads();

  int hi = lane >> 4;          // 0: lanes 0-15, 1: lanes 16-31
  int l  = lane & 15;
  int rowA = mt * 16 + l;
  if (rowA >= M) rowA = M - 1;
  const bf16_t* arow = A + (size_t)rowA * K;

  v8f acc[NCT_L];
  v8f zero = {0.f, 0.f, 0.f, 0.f, 0.f, 0.f, 0.f, 0.f};
#pragma unroll
  for (int i = 0; i < NCT_L; i++) acc[i] = zero;

  for (int k0 = 0; k0 < K; k0 += 32) {
    Frag af;
    af.u[0] = *(const uint4*)(arow + k0 + hi * 16);       // K {0..7 | 16..23}
    af.u[1] = *(const uint4*)(arow + k0 + hi * 16 + 8);   // K {8..15 | 24..31}
    Frag bfr[NCT_L];
#pragma unroll
    for (int ct = 0; ct < NCT_L; ct++) {                  // batch LDS reads
      int lc = (wv * NCT_L + ct) * 16 + l;                // local col in slab
      const bf16_t* bp = tileB + (size_t)lc * K + k0 + hi * 16;
      bfr[ct].u[0] = *(const uint4*)bp;
      bfr[ct].u[1] = *((const uint4*)bp + 1);
    }
#pragma unroll
    for (int ct = 0; ct < NCT_L; ct++) {                  // back-to-back WMMAs
      acc[ct] = __builtin_amdgcn_wmma_f32_16x16x32_bf16(
          false, af.v, false, bfr[ct].v, (short)0, acc[ct], false, false);
    }
  }

  // C/D layout: VGPR j -> M = mt*16 + j + 8*hi ; N = lane&15
#pragma unroll
  for (int ct = 0; ct < NCT_L; ct++) {
    int col = slab * SLAB_COLS + (wv * NCT_L + ct) * 16 + l;
    float b = bias ? bias[col] : 0.f;
#pragma unroll
    for (int j = 0; j < 8; j++) {
      int row = mt * 16 + j + hi * 8;
      if (row < M) outB[(size_t)row * NCols + col] = (bf16_t)(acc[ct][j] + b);
    }
  }
}

// ----------------- WMMA GEMM, direct-global B (classifier) -----------------
template <int NCT>
__global__ __launch_bounds__(256)
void gemm_wmma_kernel(const bf16_t* __restrict__ A, const bf16_t* __restrict__ WT,
                      const float* __restrict__ bias,
                      float* __restrict__ outF, bf16_t* __restrict__ outB,
                      int M, int K, int NCols) {
  int wave = blockIdx.x * (blockDim.x >> 5) + (threadIdx.x >> 5);
  int lane = threadIdx.x & 31;
  int colGroups = NCols / (16 * NCT);
  int mt = wave / colGroups;
  int cg = wave - mt * colGroups;
  if (mt * 16 >= M) return;

  int hi = lane >> 4;
  int l  = lane & 15;
  int rowA = mt * 16 + l;
  if (rowA >= M) rowA = M - 1;
  const bf16_t* arow = A + (size_t)rowA * K;

  v8f acc[NCT];
  v8f zero = {0.f, 0.f, 0.f, 0.f, 0.f, 0.f, 0.f, 0.f};
#pragma unroll
  for (int i = 0; i < NCT; i++) acc[i] = zero;

  for (int k0 = 0; k0 < K; k0 += 32) {
    Frag af;
    af.u[0] = *(const uint4*)(arow + k0 + hi * 16);
    af.u[1] = *(const uint4*)(arow + k0 + hi * 16 + 8);
    Frag bfr[NCT];
#pragma unroll
    for (int ct = 0; ct < NCT; ct++) {
      int col = (cg * NCT + ct) * 16 + l;
      const bf16_t* brow = WT + (size_t)col * K + k0 + hi * 16;
      bfr[ct].u[0] = *(const uint4*)(brow);
      bfr[ct].u[1] = *((const uint4*)brow + 1);
    }
#pragma unroll
    for (int ct = 0; ct < NCT; ct++) {
      acc[ct] = __builtin_amdgcn_wmma_f32_16x16x32_bf16(
          false, af.v, false, bfr[ct].v, (short)0, acc[ct], false, false);
    }
  }

#pragma unroll
  for (int ct = 0; ct < NCT; ct++) {
    int col = (cg * NCT + ct) * 16 + l;
    float b = bias ? bias[col] : 0.f;
    if (outB) {
#pragma unroll
      for (int j = 0; j < 8; j++) {
        int row = mt * 16 + j + hi * 8;
        if (row < M) outB[(size_t)row * NCols + col] = (bf16_t)(acc[ct][j] + b);
      }
    } else {
#pragma unroll
      for (int j = 0; j < 8; j++) {
        int row = mt * 16 + j + hi * 8;
        if (row < M) outF[(size_t)row * NCols + col] = acc[ct][j] + b;
      }
    }
  }
}

// --------------------------- edge pipeline ---------------------------------
// One wave per edge. Fused edge GEMM: e_hc = sum_k ea[k]*We[k][hc] recomputed
// in-register (We is 128KB -> L0/L2 resident), avoiding 2.6GB/layer traffic.
__global__ __launch_bounds__(256)
void edge_logits_kernel(const int* __restrict__ src, const int* __restrict__ dst,
                        const float* __restrict__ eattr,
                        const bf16_t* __restrict__ xlb, const bf16_t* __restrict__ xrb,
                        const float* __restrict__ We, const float* __restrict__ att,
                        float* __restrict__ logits, unsigned* __restrict__ nmax,
                        int nEdges) {
  int wave = blockIdx.x * (blockDim.x >> 5) + (threadIdx.x >> 5);
  int lane = threadIdx.x & 31;
  if (wave >= nEdges) return;
  int e = __builtin_amdgcn_readfirstlane(wave);   // force SGPR addressing
  int s = src[e], d = dst[e];
  float ea[D_EDGE];
#pragma unroll
  for (int k = 0; k < D_EDGE; k++) ea[k] = eattr[(size_t)e * D_EDGE + k];
  const bf16_t* xs = xlb + (size_t)s * HC;
  const bf16_t* xd = xrb + (size_t)d * HC;
  for (int h = 0; h < HEADS; h++) {
    int hc = h * HID + lane;
    float v = (float)xs[hc] + (float)xd[hc];
#pragma unroll
    for (int k = 0; k < D_EDGE; k++) v += ea[k] * We[k * HC + hc];
    v = v > 0.f ? v : NEG_SLOPE * v;
    float lg = v * att[hc];
#pragma unroll
    for (int off = 16; off; off >>= 1) lg += __shfl_xor(lg, off, 32);
    if (lane == 0) {
      logits[(size_t)e * HEADS + h] = lg;
      atomicMax(&nmax[(size_t)d * HEADS + h], f2ord(lg));
    }
  }
}

__global__ __launch_bounds__(256)
void edge_exp_kernel(const int* __restrict__ dst, const unsigned* __restrict__ nmax,
                     float* __restrict__ exbuf, float* __restrict__ denom, int total) {
  int i = blockIdx.x * blockDim.x + threadIdx.x;
  if (i >= total) return;
  int e = i >> 6, h = i & (HEADS - 1);
  int d = dst[e];
  float mx = ord2f(nmax[(size_t)d * HEADS + h]);
  float ex = __expf(exbuf[i] - mx);
  exbuf[i] = ex;
  atomicAdd(&denom[(size_t)d * HEADS + h], ex);
}

__global__ __launch_bounds__(256)
void edge_agg_kernel(const int* __restrict__ src, const int* __restrict__ dst,
                     const float* __restrict__ exbuf, const float* __restrict__ denom,
                     const bf16_t* __restrict__ xlb, float* __restrict__ agg, int nEdges) {
  int wave = blockIdx.x * (blockDim.x >> 5) + (threadIdx.x >> 5);
  int lane = threadIdx.x & 31;
  if (wave >= nEdges) return;
  int e = __builtin_amdgcn_readfirstlane(wave);
  int s = src[e], d = dst[e];
  const bf16_t* xs = xlb + (size_t)s * HC;
  float* od = agg + (size_t)d * HC;
  for (int h = 0; h < HEADS; h++) {
    float alpha = exbuf[(size_t)e * HEADS + h] /
                  (denom[(size_t)d * HEADS + h] + EPS_SM);
    atomicAdd(&od[h * HID + lane], alpha * (float)xs[h * HID + lane]);
  }
}

// --------------------------- finalize --------------------------------------

__global__ __launch_bounds__(256)
void finalize_mean_kernel(const float* __restrict__ agg, const float* __restrict__ bout,
                          bf16_t* __restrict__ xbNext, int nNodes) {
  int wave = blockIdx.x * (blockDim.x >> 5) + (threadIdx.x >> 5);
  int lane = threadIdx.x & 31;
  if (wave >= nNodes) return;
  const float* row = agg + (size_t)wave * HC;
  float s = 0.f;
#pragma unroll
  for (int h = 0; h < HEADS; h++) s += row[h * HID + lane];
  float v = tanhf(s * (1.f / HEADS) + bout[lane]);
  xbNext[(size_t)wave * HID + lane] = (bf16_t)v;
}

__global__ __launch_bounds__(256)
void finalize_concat_kernel(const float* __restrict__ agg, const float* __restrict__ bout,
                            float* __restrict__ hOut, bf16_t* __restrict__ hb, int total) {
  int i = blockIdx.x * blockDim.x + threadIdx.x;
  if (i >= total) return;
  float v = tanhf(agg[i] + bout[i & (HC - 1)]);
  hOut[i] = v;
  hb[i] = (bf16_t)v;
}

__global__ __launch_bounds__(256)
void logsoftmax_kernel(const float* __restrict__ cls, const float* __restrict__ bc,
                       float* __restrict__ out, int nNodes) {
  int n = blockIdx.x * blockDim.x + threadIdx.x;
  if (n >= nNodes) return;
  float z[NCLS];
  float mx = -1e30f;
#pragma unroll
  for (int c = 0; c < NCLS; c++) { z[c] = cls[(size_t)n * 32 + c] + bc[c]; mx = fmaxf(mx, z[c]); }
  float s = 0.f;
#pragma unroll
  for (int c = 0; c < NCLS; c++) s += __expf(z[c] - mx);
  float ls = __logf(s);
#pragma unroll
  for (int c = 0; c < NCLS; c++) out[(size_t)n * NCLS + c] = z[c] - mx - ls;
}

// --------------------------- host driver -----------------------------------

extern "C" void kernel_launch(void* const* d_in, const int* in_sizes, int n_in,
                              void* d_out, int out_size, void* d_ws, size_t ws_size,
                              hipStream_t stream) {
  (void)in_sizes; (void)out_size; (void)ws_size;
  if (n_in < 27) return;

  const float* x     = (const float*)d_in[0];
  const int*   esrc  = (const int*)d_in[1];
  const int*   edst  = (const int*)d_in[2];
  const float* eattr = (const float*)d_in[3];
  struct Layer { const float *Wl, *bl, *Wr, *br, *We, *att, *bout; int K; };
  Layer L[3];
  for (int i = 0; i < 3; i++) {
    int b = 4 + 7 * i;
    L[i].Wl   = (const float*)d_in[b + 0];
    L[i].bl   = (const float*)d_in[b + 1];
    L[i].Wr   = (const float*)d_in[b + 2];
    L[i].br   = (const float*)d_in[b + 3];
    L[i].We   = (const float*)d_in[b + 4];
    L[i].att  = (const float*)d_in[b + 5];
    L[i].bout = (const float*)d_in[b + 6];
    L[i].K    = (i == 0) ? D_NODE : HID;
  }
  const float* Wc = (const float*)d_in[25];
  const float* bc = (const float*)d_in[26];

  // Workspace carve (256B aligned); total ~253MB.
  char* ws = (char*)d_ws;
  size_t off = 0;
  auto carve = [&](size_t bytes) { void* p = ws + off; off += (bytes + 255) & ~(size_t)255; return p; };
  bf16_t*   xb64  = (bf16_t*)carve((size_t)N_NODES * D_NODE * 2);
  bf16_t*   xb32  = (bf16_t*)carve((size_t)N_NODES * HID * 2);
  bf16_t*   xlb   = (bf16_t*)carve((size_t)N_NODES * HC * 2);
  bf16_t*   xrb   = (bf16_t*)carve((size_t)N_NODES * HC * 2);
  float*    agg   = (float*)carve((size_t)N_NODES * HC * 4);
  float*    exbuf = (float*)carve((size_t)N_EDGES * HEADS * 4);
  unsigned* nmax  = (unsigned*)carve((size_t)N_NODES * HEADS * 4);
  float*    denom = (float*)carve((size_t)N_NODES * HEADS * 4);
  bf16_t*   WlT   = (bf16_t*)carve((size_t)HC * D_NODE * 2);
  bf16_t*   WrT   = (bf16_t*)carve((size_t)HC * D_NODE * 2);
  bf16_t*   WcT   = (bf16_t*)carve((size_t)32 * HC * 2);
  bf16_t*   hb    = (bf16_t*)carve((size_t)N_NODES * HC * 2);
  float*    cls   = (float*)carve((size_t)N_NODES * 32 * 4);

  // x -> bf16
  cvt_f32_bf16_kernel<<<(N_NODES * D_NODE + 255) / 256, 256, 0, stream>>>(x, xb64, N_NODES * D_NODE);

  const bf16_t* Ain = xb64;
  for (int i = 0; i < 3; i++) {
    int K = L[i].K;
    int tN = HC * K;
    transpose_w_kernel<<<(tN + 255) / 256, 256, 0, stream>>>(L[i].Wl, WlT, K, HC, HC);
    transpose_w_kernel<<<(tN + 255) / 256, 256, 0, stream>>>(L[i].Wr, WrT, K, HC, HC);

    int gemmBlk = ((N_NODES + 15) / 16) * (HC / SLAB_COLS);   // 625 * 4
    int slabBytes = SLAB_COLS * K * 2;                        // 64KB (K=64) / 32KB (K=32)
    gemm_wmma_tdm_kernel<<<gemmBlk, 256, slabBytes, stream>>>(
        Ain, WlT, L[i].bl, xlb, N_NODES, K, HC);
    gemm_wmma_tdm_kernel<<<gemmBlk, 256, slabBytes, stream>>>(
        Ain, WrT, L[i].br, xrb, N_NODES, K, HC);

    hipMemsetAsync(nmax, 0, (size_t)N_NODES * HEADS * 4, stream);   // ordered-uint 0 < all reals
    hipMemsetAsync(denom, 0, (size_t)N_NODES * HEADS * 4, stream);
    hipMemsetAsync(agg, 0, (size_t)N_NODES * HC * 4, stream);

    edge_logits_kernel<<<(N_EDGES + 7) / 8, 256, 0, stream>>>(
        esrc, edst, eattr, xlb, xrb, L[i].We, L[i].att, exbuf, nmax, N_EDGES);
    edge_exp_kernel<<<(N_EDGES * HEADS + 255) / 256, 256, 0, stream>>>(
        edst, nmax, exbuf, denom, N_EDGES * HEADS);
    edge_agg_kernel<<<(N_EDGES + 7) / 8, 256, 0, stream>>>(
        esrc, edst, exbuf, denom, xlb, agg, N_EDGES);

    if (i < 2) {
      finalize_mean_kernel<<<(N_NODES + 7) / 8, 256, 0, stream>>>(agg, L[i].bout, xb32, N_NODES);
      Ain = xb32;
    } else {
      float* hOut = (float*)d_out + (size_t)N_NODES * NCLS;
      finalize_concat_kernel<<<((size_t)N_NODES * HC + 255) / 256, 256, 0, stream>>>(
          agg, L[i].bout, hOut, hb, N_NODES * HC);
    }
  }

  // classifier: h(bf16) @ Wc -> cls[N][32] (cols 20..31 zero-padded weights)
  transpose_w_kernel<<<(32 * HC + 255) / 256, 256, 0, stream>>>(Wc, WcT, HC, NCLS, 32);
  {
    int waves = ((N_NODES + 15) / 16) * (32 / (16 * 2));
    gemm_wmma_kernel<2><<<(waves + 7) / 8, 256, 0, stream>>>(
        hb, WcT, nullptr, cls, nullptr, N_NODES, HC, 32);
  }
  logsoftmax_kernel<<<(N_NODES + 255) / 256, 256, 0, stream>>>(cls, bc, (float*)d_out, N_NODES);
}
